// MultiHeadSelfAttention_8933531976130
// MI455X (gfx1250) — matrix operand order repaired and checked by reading, not compiled, not run on hardware
//
#include <hip/hip_runtime.h>
#include <hip/hip_bf16.h>
#include <stdint.h>

// ---------------------------------------------------------------------------
// MI455X (gfx1250) fused MHA: bf16 WMMA GEMMs + flash-attention core.
// Round 2: async global->LDS staging (ASYNCcnt path) for row-major tiles.
// ---------------------------------------------------------------------------

#define USE_ASYNC_STAGING 1

typedef __attribute__((ext_vector_type(16))) __bf16 bf16x16;
typedef __attribute__((ext_vector_type(8)))  float  v8f;

static constexpr int S_LEN  = 2048;
static constexpr int DMODEL = 1024;
static constexpr int NHEAD  = 16;
static constexpr int DK     = 64;
static constexpr int BATCH  = 2;
static constexpr int MTOT   = BATCH * S_LEN;  // 4096 rows

union FragA { uint32_t u[8]; bf16x16 v; };
union AccU  { float    f[8]; v8f     v; };

__device__ __forceinline__ uint16_t f32_bf16(float x) {
    uint32_t u = __float_as_uint(x);
    uint32_t r = u + 0x7FFFu + ((u >> 16) & 1u);   // round-to-nearest-even
    return (uint16_t)(r >> 16);
}

// Raw LDS byte offset of a __shared__ pointer: generic LDS addresses carry the
// LDS offset in their low 32 bits (aperture mapping, ISA 10.2).
__device__ __forceinline__ uint32_t lds_offset(const void* p) {
    return (uint32_t)(uintptr_t)p;
}

#if USE_ASYNC_STAGING
// 16B/lane async copy global -> LDS, tracked by ASYNCcnt (ISA 10.7/15.18).
__device__ __forceinline__ void async_copy_b128(uint32_t lds_byte, uint64_t gaddr) {
    asm volatile("global_load_async_to_lds_b128 %0, %1, off"
                 :: "v"(lds_byte), "v"(gaddr) : "memory");
}
__device__ __forceinline__ void wait_async0() {
    asm volatile("s_wait_asynccnt 0x0" ::: "memory");
}
#endif

// ---------------------------------------------------------------------------
// Stage 1: fp32 -> bf16 conversion (grid-stride).
// ---------------------------------------------------------------------------
__global__ void conv_f32_to_bf16(const float* __restrict__ src,
                                 uint16_t* __restrict__ dst, int n) {
    int i = blockIdx.x * blockDim.x + threadIdx.x;
    int stride = gridDim.x * blockDim.x;
    for (; i < n; i += stride) dst[i] = f32_bf16(src[i]);
}

// ---------------------------------------------------------------------------
// Stage 2/4: C[M,1024] = A[M,1024](bf16) x W[1024,1024](bf16) + bias.
// Block: 256 thr = 8 waves; tile 128(M) x 128(N); K-step 32.
// ---------------------------------------------------------------------------
template <bool BF16_OUT>
__global__ __launch_bounds__(256) void gemm_bf16(
    const uint16_t* __restrict__ A,    // [M][1024] bf16 row-major
    const uint16_t* __restrict__ W,    // [1024][1024] bf16 (K x N) row-major
    const float*    __restrict__ bias, // [1024] fp32
    void*           __restrict__ Cout) // bf16 or fp32 [M][1024]
{
    __shared__ uint32_t aS[128 * 16];  // A tile [128 rows][32 elems] (64B/row)
    __shared__ uint32_t bS[128 * 16];  // W tile transposed [128 n][32 k]

    const int tid  = threadIdx.x;
    const int wave = tid >> 5;
    const int lane = tid & 31;
    const int hh   = lane >> 4;        // 16-lane half id
    const int l16  = lane & 15;

    const int n0 = blockIdx.x * 128;
    const int m0 = blockIdx.y * 128;

    const uint32_t aS_base = lds_offset(&aS[0]);

    AccU acc[8];
    for (int i = 0; i < 8; ++i)
        for (int r = 0; r < 8; ++r) acc[i].f[r] = 0.f;

    for (int k0 = 0; k0 < DMODEL; k0 += 32) {
#if USE_ASYNC_STAGING
        // Stage A tile with async DMA: 8KB = 256 lanes x 2 x 16B.
        for (int j = 0; j < 2; ++j) {
            int byte = (tid + j * 256) * 16;
            int row  = byte >> 6;       // 64B per row
            int colB = byte & 63;
            uint64_t g = (uint64_t)(uintptr_t)((const char*)A +
                ((size_t)(m0 + row) * DMODEL + k0) * 2 + colB);
            async_copy_b128(aS_base + byte, g);
        }
#else
        for (int i = 0; i < 8; ++i) {
            int u = tid + i * 256;
            int row = u >> 4, cu = u & 15;
            aS[u] = *(const uint32_t*)(A + (size_t)(m0 + row) * DMODEL + k0 + cu * 2);
        }
#endif
        // Stage W tile transposed into [n][k] (must pass through VGPRs).
        for (int i = 0; i < 8; ++i) {
            int u  = tid + i * 256;
            int kr = u >> 6;           // 64 u32 per k-row (128 n values)
            int np = (u & 63) * 2;     // n pair
            uint32_t w = *(const uint32_t*)(W + (size_t)(k0 + kr) * DMODEL + n0 + np);
            uint16_t* b16 = (uint16_t*)bS;
            b16[(np + 0) * 32 + kr] = (uint16_t)(w & 0xFFFFu);
            b16[(np + 1) * 32 + kr] = (uint16_t)(w >> 16);
        }
#if USE_ASYNC_STAGING
        wait_async0();
#endif
        __syncthreads();

        // A fragment: lane%16 = row; half selects K sub-bands.
        FragA af;
        {
            int row = wave * 16 + l16;
            for (int i = 0; i < 4; ++i) af.u[i]     = aS[row * 16 + hh * 4 + i];
            for (int i = 0; i < 4; ++i) af.u[4 + i] = aS[row * 16 + 8 + hh * 4 + i];
        }
        for (int nc = 0; nc < 8; ++nc) {
            FragA bfg;
            int n = nc * 16 + l16;
            for (int i = 0; i < 8; ++i) bfg.u[i] = bS[n * 16 + (i >> 2) * 8 + hh * 4 + (i & 3)];
            acc[nc].v = __builtin_amdgcn_wmma_f32_16x16x32_bf16(
                false, af.v, false, bfg.v, (short)0, acc[nc].v, false, false);
        }
        __syncthreads();
    }

    // Epilogue: C layout -> row M = r + hh*8 (+wave*16), col N = lane%16.
    for (int nc = 0; nc < 8; ++nc) {
        int col = n0 + nc * 16 + l16;
        float bv = bias[col];
        for (int r = 0; r < 8; ++r) {
            int m = m0 + wave * 16 + hh * 8 + r;
            float v = acc[nc].f[r] + bv;
            if (BF16_OUT)
                ((uint16_t*)Cout)[(size_t)m * DMODEL + col] = f32_bf16(v);
            else
                ((float*)Cout)[(size_t)m * DMODEL + col] = v;
        }
    }
}

// ---------------------------------------------------------------------------
// Stage 3: flash attention. Grid (S/128, H, B); 256 thr = 8 waves.
// ---------------------------------------------------------------------------
__global__ __launch_bounds__(256) void flash_attn(
    const uint16_t* __restrict__ Q,    // [B][S][H*DK] bf16
    const uint16_t* __restrict__ Kd,   // same layout
    const uint16_t* __restrict__ V,    // same layout
    uint16_t* __restrict__ O)          // [B][S][H*DK] bf16
{
    __shared__ uint32_t kS[1024];   // K tile  [32 t][64 d]  bf16 (128B/row)
    __shared__ uint32_t vS[1024];   // V tileT [64 d][32 t]  bf16
    __shared__ uint32_t pS[2048];   // per-wave P scratch [16][32] bf16

    const int tid  = threadIdx.x;
    const int wave = tid >> 5;
    const int lane = tid & 31;
    const int hh   = lane >> 4;
    const int l16  = lane & 15;

    const int b  = blockIdx.z;
    const int h  = blockIdx.y;
    const int s0 = blockIdx.x * 128;

    const size_t headOff = (size_t)b * S_LEN * DMODEL + (size_t)h * DK;
    const uint32_t kS_base = lds_offset(&kS[0]);

    // Q fragments for both Dk-chunks (K=32 each), loaded once.
    FragA qf[2];
    {
        int srow = s0 + wave * 16 + l16;
        const uint16_t* qp = Q + headOff + (size_t)srow * DMODEL;
        for (int c = 0; c < 2; ++c) {
            const uint16_t* base = qp + c * 32;
            for (int i = 0; i < 4; ++i)
                qf[c].u[i]     = *(const uint32_t*)(base + hh * 8 + 2 * i);
            for (int i = 0; i < 4; ++i)
                qf[c].u[4 + i] = *(const uint32_t*)(base + 16 + hh * 8 + 2 * i);
        }
    }

    float m_r[8], l_r[8];
    AccU  ao[4];
    for (int r = 0; r < 8; ++r) { m_r[r] = -1e30f; l_r[r] = 0.f; }
    for (int i = 0; i < 4; ++i)
        for (int r = 0; r < 8; ++r) ao[i].f[r] = 0.f;

    const float sm_scale = 0.125f;  // 1/sqrt(64)

    for (int t0 = 0; t0 < S_LEN; t0 += 32) {
#if USE_ASYNC_STAGING
        // Stage K tile [32][64] via async DMA: 4KB = 256 lanes x 16B.
        {
            int byte = tid * 16;
            int tr   = byte >> 7;       // 128B per key row
            int off  = byte & 127;
            uint64_t g = (uint64_t)(uintptr_t)((const char*)Kd +
                (headOff + (size_t)(t0 + tr) * DMODEL) * 2 + off);
            async_copy_b128(kS_base + byte, g);
        }
#else
        for (int i = 0; i < 4; ++i) {
            int u  = tid + i * 256;
            int tr = u >> 5, du = u & 31;
            kS[u] = *(const uint32_t*)(Kd + headOff + (size_t)(t0 + tr) * DMODEL + du * 2);
        }
#endif
        // Stage V transposed [d][t] (register transpose).
        for (int i = 0; i < 8; ++i) {
            int e  = tid + i * 256;
            int tr = e >> 6, dd = e & 63;
            ((uint16_t*)vS)[dd * 32 + tr] = V[headOff + (size_t)(t0 + tr) * DMODEL + dd];
        }
        // Prefetch next K tile into cache hierarchy.
        if (t0 + 32 < S_LEN)
            __builtin_prefetch(Kd + headOff + (size_t)(t0 + 32) * DMODEL + lane * 32, 0, 0);
#if USE_ASYNC_STAGING
        wait_async0();
#endif
        __syncthreads();

        // Scores: 16x32 per wave = 2 N-chunks x 2 K-chunks of WMMA.
        AccU sc[2];
        for (int j = 0; j < 2; ++j)
            for (int r = 0; r < 8; ++r) sc[j].f[r] = 0.f;
        for (int j = 0; j < 2; ++j) {
            for (int c = 0; c < 2; ++c) {
                FragA kf;
                int n = j * 16 + l16;
                for (int i = 0; i < 8; ++i)
                    kf.u[i] = kS[n * 32 + c * 16 + (i >> 2) * 8 + hh * 4 + (i & 3)];
                sc[j].v = __builtin_amdgcn_wmma_f32_16x16x32_bf16(
                    false, qf[c].v, false, kf.v, (short)0, sc[j].v, false, false);
            }
        }

        // Online softmax. Row M = hh*8 + r lives on 16 lanes (N = lane%16).
        uint16_t* pw = (uint16_t*)(pS + wave * 256);
        for (int r = 0; r < 8; ++r) {
            float a0 = sc[0].f[r] * sm_scale;
            float a1 = sc[1].f[r] * sm_scale;
            float mx = fmaxf(a0, a1);
            for (int off = 1; off < 16; off <<= 1)
                mx = fmaxf(mx, __shfl_xor(mx, off, 32));
            float mnew  = fmaxf(m_r[r], mx);
            float scale = __expf(m_r[r] - mnew);
            float p0 = __expf(a0 - mnew);
            float p1 = __expf(a1 - mnew);
            float rs = p0 + p1;
            for (int off = 1; off < 16; off <<= 1)
                rs += __shfl_xor(rs, off, 32);
            l_r[r] = l_r[r] * scale + rs;
            m_r[r] = mnew;
            for (int nc = 0; nc < 4; ++nc) ao[nc].f[r] *= scale;
            int prow = hh * 8 + r;
            pw[prow * 32 + l16]      = f32_bf16(p0);
            pw[prow * 32 + 16 + l16] = f32_bf16(p1);
        }
        // Same-wave LDS ops complete in order (DScnt): safe to regather.

        // P (A-operand) x V: 4 WMMAs over Dk chunks.
        FragA pf;
        {
            const uint32_t* pb = pS + wave * 256;   // [16 rows][16 u32]
            int row = l16;
            for (int i = 0; i < 4; ++i) pf.u[i]     = pb[row * 16 + hh * 4 + i];
            for (int i = 0; i < 4; ++i) pf.u[4 + i] = pb[row * 16 + 8 + hh * 4 + i];
        }
        for (int nc = 0; nc < 4; ++nc) {
            FragA vf;
            int n = nc * 16 + l16;
            for (int i = 0; i < 8; ++i) vf.u[i] = vS[n * 16 + (i >> 2) * 8 + hh * 4 + (i & 3)];
            ao[nc].v = __builtin_amdgcn_wmma_f32_16x16x32_bf16(
                false, pf.v, false, vf.v, (short)0, ao[nc].v, false, false);
        }
        __syncthreads();
    }

    // Normalize and store bf16 attn output in [B][S][H*DK].
    {
        int srow = s0 + wave * 16 + hh * 8;
        for (int nc = 0; nc < 4; ++nc) {
            for (int r = 0; r < 8; ++r) {
                float v = ao[nc].f[r] / l_r[r];
                O[(size_t)b * S_LEN * DMODEL + (size_t)(srow + r) * DMODEL +
                  h * DK + nc * 16 + l16] = f32_bf16(v);
            }
        }
    }
}

// ---------------------------------------------------------------------------
// Host-side launcher.
// ---------------------------------------------------------------------------
extern "C" void kernel_launch(void* const* d_in, const int* in_sizes, int n_in,
                              void* d_out, int out_size, void* d_ws, size_t ws_size,
                              hipStream_t stream) {
    const float* x   = (const float*)d_in[0];
    const float* w_q = (const float*)d_in[1];
    const float* b_q = (const float*)d_in[2];
    const float* w_k = (const float*)d_in[3];
    const float* b_k = (const float*)d_in[4];
    const float* w_v = (const float*)d_in[5];
    const float* b_v = (const float*)d_in[6];
    const float* w_o = (const float*)d_in[7];
    const float* b_o = (const float*)d_in[8];
    float* out = (float*)d_out;

    char* ws = (char*)d_ws;
    const size_t MB = 1u << 20;
    uint16_t* x_bf  = (uint16_t*)(ws + 0 * MB);   // 8 MB
    uint16_t* wq_bf = (uint16_t*)(ws + 8 * MB);   // 2 MB
    uint16_t* wk_bf = (uint16_t*)(ws + 10 * MB);  // 2 MB
    uint16_t* wv_bf = (uint16_t*)(ws + 12 * MB);  // 2 MB
    uint16_t* wo_bf = (uint16_t*)(ws + 14 * MB);  // 2 MB
    uint16_t* Qb    = (uint16_t*)(ws + 16 * MB);  // 8 MB
    uint16_t* Kb    = (uint16_t*)(ws + 24 * MB);  // 8 MB
    uint16_t* Vb    = (uint16_t*)(ws + 32 * MB);  // 8 MB
    uint16_t* Ab    = (uint16_t*)(ws + 40 * MB);  // 8 MB

    // Stage 1: convert activations + weights to bf16.
    conv_f32_to_bf16<<<2048, 256, 0, stream>>>(x,   x_bf,  MTOT * DMODEL);
    conv_f32_to_bf16<<<1024, 256, 0, stream>>>(w_q, wq_bf, DMODEL * DMODEL);
    conv_f32_to_bf16<<<1024, 256, 0, stream>>>(w_k, wk_bf, DMODEL * DMODEL);
    conv_f32_to_bf16<<<1024, 256, 0, stream>>>(w_v, wv_bf, DMODEL * DMODEL);
    conv_f32_to_bf16<<<1024, 256, 0, stream>>>(w_o, wo_bf, DMODEL * DMODEL);

    // Stage 2: QKV projections (bf16 out).
    dim3 gg(DMODEL / 128, MTOT / 128);  // (8, 32)
    gemm_bf16<true><<<gg, 256, 0, stream>>>(x_bf, wq_bf, b_q, Qb);
    gemm_bf16<true><<<gg, 256, 0, stream>>>(x_bf, wk_bf, b_k, Kb);
    gemm_bf16<true><<<gg, 256, 0, stream>>>(x_bf, wv_bf, b_v, Vb);

    // Stage 3: fused flash attention.
    flash_attn<<<dim3(S_LEN / 128, NHEAD, BATCH), 256, 0, stream>>>(Qb, Kb, Vb, Ab);

    // Stage 4: output projection (fp32 out + bias).
    gemm_bf16<false><<<gg, 256, 0, stream>>>(Ab, wo_bf, b_o, out);
}
